// SafePolicyWrapper_84164179133328
// MI455X (gfx1250) — matrix operand order repaired and checked by reading, not compiled, not run on hardware
//
#include <hip/hip_runtime.h>
#include <hip/hip_bf16.h>

typedef __attribute__((ext_vector_type(16))) __bf16         v16bf;
typedef __attribute__((ext_vector_type(8)))  unsigned int   v8ui;
typedef __attribute__((ext_vector_type(4)))  unsigned int   v4ui;
typedef __attribute__((ext_vector_type(8)))  float          v8f;
typedef __attribute__((ext_vector_type(4)))  float          v4f;

#define DIM    128
#define CLS    43
#define CPAD   48
#define THRESH 0.6f
#define WS_BYTES (CPAD * DIM * 2)   // 12288 B: bf16 transposed W

// ---- helpers -------------------------------------------------------------

// Pack two f32 -> one dword of two bf16 (round-half-up + v_perm_b32 truncate).
__device__ __forceinline__ unsigned pack_bf16(float lo, float hi) {
    unsigned ul = __builtin_bit_cast(unsigned, lo) + 0x8000u;
    unsigned uh = __builtin_bit_cast(unsigned, hi) + 0x8000u;
    return __builtin_amdgcn_perm(uh, ul, 0x07060302u);  // {uh[31:16], ul[31:16]}
}

// f32 -> bf16 bits, RNE (prep path only)
__device__ __forceinline__ unsigned short f2bf(float f) {
    unsigned u = __builtin_bit_cast(unsigned, f);
    unsigned r = u + 0x7FFFu + ((u >> 16) & 1u);
    return (unsigned short)(r >> 16);
}

// Immediate-mode ds_swizzle XOR reductions (group-of-32, and_mask=0x1F)
template <int IMM> __device__ __forceinline__ float swzf(float v) {
    return __builtin_bit_cast(float,
        __builtin_amdgcn_ds_swizzle(__builtin_bit_cast(int, v), IMM));
}
template <int IMM> __device__ __forceinline__ unsigned swzu(unsigned v) {
    return (unsigned)__builtin_amdgcn_ds_swizzle((int)v, IMM);
}
#define SWZ_X1 0x041F
#define SWZ_X2 0x081F
#define SWZ_X4 0x101F
#define SWZ_X8 0x201F

// Order-preserving f32->u32 key; low 6 bits carry (63-col) for tie->lowest col
__device__ __forceinline__ unsigned mkkey(float v, int col) {
    unsigned u = __builtin_bit_cast(unsigned, v);
    unsigned k = ((int)u < 0) ? ~u : (u | 0x80000000u);
    return (k & 0xFFFFFFC0u) | (unsigned)(63 - col);
}

// ---- prep kernel: W -> bf16, transposed [CPAD][DIM], zero-padded ---------

__global__ __launch_bounds__(256) void prep_W(
    const float* __restrict__ W, unsigned short* __restrict__ ws)
{
    const int tid = threadIdx.x;
    for (int idx = tid; idx < DIM * CLS; idx += 256) {
        int k   = idx / CLS;
        int col = idx - k * CLS;
        ws[col * DIM + k] = f2bf(W[idx]);
    }
    for (int idx = tid; idx < (CPAD - CLS) * DIM; idx += 256)
        ws[CLS * DIM + idx] = 0;
}

// ---- main kernel ---------------------------------------------------------

template <bool STAGE_WS>
__global__ __launch_bounds__(256) void safe_policy_fused(
    const float* __restrict__ x, const float* __restrict__ W,
    const unsigned short* __restrict__ ws,
    const float* __restrict__ bias, float* __restrict__ out)
{
    __shared__ unsigned short WlT[CPAD * DIM];   // [col][k], bf16 bits

    const int tid = threadIdx.x;
    if (STAGE_WS) {
        // plain 48B/thread block copy of the prepped image
        const v4ui* src = (const v4ui*)ws;
        v4ui*       dst = (v4ui*)WlT;
        #pragma unroll
        for (int i = 0; i < 3; ++i) dst[tid + 256 * i] = src[tid + 256 * i];
    } else {
        // fallback: convert + transpose in-block
        for (int idx = tid; idx < DIM * CLS; idx += 256) {
            int k   = idx / CLS;
            int col = idx - k * CLS;
            WlT[col * DIM + k] = f2bf(W[idx]);
        }
        for (int idx = tid; idx < (CPAD - CLS) * DIM; idx += 256)
            WlT[CLS * DIM + idx] = 0;
    }
    __syncthreads();

    const int lane    = tid & 31;
    const int wave    = tid >> 5;
    const int rowBase = (blockIdx.x * 8 + wave) * 16;
    const int half    = lane >> 4;     // 0: lanes 0-15, 1: lanes 16-31
    const int l16     = lane & 15;

    v8f acc0 = {}, acc1 = {}, acc2 = {};

    const float* xrow = x + (size_t)(rowBase + l16) * DIM;
    const int kbA = half * 8;    // 16-bit A layout: halves own K 0-7/16-23 vs 8-15/24-31
    const int kbB = half * 16;   // unused: B loads index with kbA, same K split as A

    #pragma unroll
    for (int k0 = 0; k0 < DIM; k0 += 32) {
        // ---- A fragment (16x32 bf16): two aligned 32B runs per lane ----
        const v4f* p0 = (const v4f*)(xrow + k0 + kbA);        // K = kbA+0..7
        const v4f* p1 = (const v4f*)(xrow + k0 + kbA + 16);   // K = kbA+16..23
        v4f f0 = p0[0], f1 = p0[1], f2 = p1[0], f3 = p1[1];
        v8ui au;
        au[0] = pack_bf16(f0[0], f0[1]);  au[1] = pack_bf16(f0[2], f0[3]);
        au[2] = pack_bf16(f1[0], f1[1]);  au[3] = pack_bf16(f1[2], f1[3]);
        au[4] = pack_bf16(f2[0], f2[1]);  au[5] = pack_bf16(f2[2], f2[3]);
        au[6] = pack_bf16(f3[0], f3[1]);  au[7] = pack_bf16(f3[2], f3[3]);
        v16bf a = __builtin_bit_cast(v16bf, au);

        // ---- B fragments (32x16 bf16) from LDS: K = kbA+0..7 and kbA+16..23 per lane ----
        #pragma unroll
        for (int t = 0; t < 3; ++t) {
            const int col = t * 16 + l16;
            const v4ui* bp = (const v4ui*)&WlT[col * DIM + k0 + kbA];
            v4ui blo = bp[0], bhi = bp[2];
            v8ui bu;
            #pragma unroll
            for (int j = 0; j < 4; ++j) { bu[j] = blo[j]; bu[4 + j] = bhi[j]; }
            v16bf bf = __builtin_bit_cast(v16bf, bu);
            if (t == 0) acc0 = __builtin_amdgcn_wmma_f32_16x16x32_bf16(false, a, false, bf, (short)0, acc0, false, false);
            if (t == 1) acc1 = __builtin_amdgcn_wmma_f32_16x16x32_bf16(false, a, false, bf, (short)0, acc1, false, false);
            if (t == 2) acc2 = __builtin_amdgcn_wmma_f32_16x16x32_bf16(false, a, false, bf, (short)0, acc2, false, false);
        }
    }

    // ---- fused epilogue: softmax entropy + argmax + policy + one-hot ----
    const bool  val2 = (32 + l16) < CLS;
    const float b0 = bias[l16];
    const float b1 = bias[16 + l16];
    const float b2 = bias[val2 ? (32 + l16) : (CLS - 1)];   // clamped, branch-free
    const float inv_logC = 1.0f / __logf((float)CLS);

    #pragma unroll
    for (int r = 0; r < 8; ++r) {
        float v0 = acc0[r] + b0;
        float v1 = acc1[r] + b1;
        float v2 = acc2[r] + b2;
        float v2m = val2 ? v2 : -3.4e38f;

        // ONE combined max/argmax reduction over sortable keys.
        // (entropy is invariant to the max offset, so the 26-bit-decoded
        //  vmax -- within 63 ulp of the true max -- is exact for the result)
        unsigned kk = mkkey(v0, l16);
        unsigned k1 = mkkey(v1, 16 + l16);
        unsigned k2 = mkkey(v2m, 32 + l16);
        kk = kk > k1 ? kk : k1;
        kk = kk > k2 ? kk : k2;
        unsigned o;
        o = swzu<SWZ_X1>(kk); kk = kk > o ? kk : o;
        o = swzu<SWZ_X2>(kk); kk = kk > o ? kk : o;
        o = swzu<SWZ_X4>(kk); kk = kk > o ? kk : o;
        o = swzu<SWZ_X8>(kk); kk = kk > o ? kk : o;
        int pred = 63 - (int)(kk & 63u);

        // decode approximate row max from the winning key
        unsigned mskd = kk & 0xFFFFFFC0u;
        unsigned um   = (kk & 0x80000000u) ? (mskd & 0x7FFFFFFFu) : ~mskd;
        float vmax = __builtin_bit_cast(float, um);

        float d0 = v0 - vmax, d1 = v1 - vmax, d2 = v2 - vmax;
        float e0 = __expf(d0), e1 = __expf(d1), e2 = val2 ? __expf(d2) : 0.0f;
        float s  = e0 + e1 + e2;
        float q  = fmaf(e0, d0, fmaf(e1, d1, val2 ? e2 * d2 : 0.0f));
        s += swzf<SWZ_X1>(s);  q += swzf<SWZ_X1>(q);
        s += swzf<SWZ_X2>(s);  q += swzf<SWZ_X2>(q);
        s += swzf<SWZ_X4>(s);  q += swzf<SWZ_X4>(q);
        s += swzf<SWZ_X8>(s);  q += swzf<SWZ_X8>(q);

        float entropy = __logf(s) - q / s;        // == -sum p*log p (shift-invariant)
        float unc     = entropy * inv_logC;

        bool high = unc > THRESH;
        if (high && pred <= 8)       pred = 1;    // speed-limit classes -> 30 km/h
        else if (high && pred <= 14) pred = 14;   // priority classes -> stop sign

        int row = rowBase + r + half * 8;
        float* orow = out + (size_t)row * CLS;
        orow[l16]      = (l16      == pred) ? 100.0f : -100.0f;
        orow[16 + l16] = (16 + l16 == pred) ? 100.0f : -100.0f;
        if (val2) orow[32 + l16] = (32 + l16 == pred) ? 100.0f : -100.0f;
    }
}

extern "C" void kernel_launch(void* const* d_in, const int* in_sizes, int n_in,
                              void* d_out, int out_size, void* d_ws, size_t ws_size,
                              hipStream_t stream) {
    (void)n_in; (void)out_size;
    const float* x    = (const float*)d_in[0];
    const float* W    = (const float*)d_in[1];
    const float* bias = (const float*)d_in[2];
    float* out = (float*)d_out;

    const int n = in_sizes[0] / DIM;          // 262144 rows
    const int rowsPerBlock = 8 * 16;          // 8 waves x 16-row tiles
    dim3 grid((n + rowsPerBlock - 1) / rowsPerBlock);
    dim3 block(256);

    if (d_ws != nullptr && ws_size >= (size_t)WS_BYTES) {
        unsigned short* ws = (unsigned short*)d_ws;
        prep_W<<<dim3(1), block, 0, stream>>>(W, ws);
        safe_policy_fused<true><<<grid, block, 0, stream>>>(x, W, ws, bias, out);
    } else {
        safe_policy_fused<false><<<grid, block, 0, stream>>>(x, W, nullptr, bias, out);
    }
}
